// AggregationBlock_62457414418604
// MI455X (gfx1250) — matrix operand order repaired and checked by reading, not compiled, not run on hardware
//
#include <hip/hip_runtime.h>
#include <math.h>

// ---------------- WMMA types ----------------
typedef __attribute__((ext_vector_type(16))) __bf16 v16bf;
typedef __attribute__((ext_vector_type(8)))  float  v8f;

// Problem constants
#define BB   8
#define NPT  2048
#define KNN  16
#define DIMC 2048
#define HH   8
#define DH   256
#define TT   128
#define HIDN 4096
#define MQ   1024
#define CT   32
#define NF1  128

// padded LDS tile stride (elements): 40 bf16 = 80 bytes -> conflict-free frag reads
#define TSTR 40

__device__ __forceinline__ float gelu_exact(float x) {
    return 0.5f * x * (1.0f + erff(x * 0.70710678118654752f));
}

// =====================================================================
// Generic fully-strided batched GEMM:  C = act(alpha*(A x W^T) + bias) [+ resid]
// A: Mr x Kd strides (a_rs, a_cs); batch offset (z/Hh)*a_bs + (z%Hh)*a_hs
// W: Nc x Kd strides (w_rs, w_cs); batch offset (z/Hh)*w_bs + (z%Hh)*w_hs
// C: Mr x Nc strides (c_rs, c_cs); batch offset (z/Hh)*c_bs + (z%Hh)*c_hs
// Block = 128 threads (4 waves). Block tile 64x64, K-step 32, staged via LDS
// in bf16; wave w computes rows [64*by+16w, +16) x 64 cols with 4 WMMAs/step.
// Kd must be a multiple of 32. OOB rows/cols are clamped (never stored).
// =====================================================================
__global__ void __launch_bounds__(128)
gemm_bf16_wmma(const float* __restrict__ A, const float* __restrict__ Wt,
               const float* __restrict__ bias, const float* __restrict__ resid,
               float* __restrict__ C,
               int Mrows, int Ncols, int Kd, int Hh,
               long a_rs, long a_cs, long a_bs, long a_hs,
               long w_rs, long w_cs, long w_bs, long w_hs,
               long c_rs, long c_cs, long c_bs, long c_hs,
               float alpha, int act)
{
    __shared__ __attribute__((aligned(16))) __bf16 As[64 * TSTR];
    __shared__ __attribute__((aligned(16))) __bf16 Bs[64 * TSTR];

    int z  = blockIdx.z;
    int zb = z / Hh, zh = z % Hh;
    long aoff = (long)zb * a_bs + (long)zh * a_hs;
    long woff = (long)zb * w_bs + (long)zh * w_hs;
    long coff = (long)zb * c_bs + (long)zh * c_hs;

    int tid  = threadIdx.x;
    int wave = tid >> 5;
    int lane = tid & 31;
    int hi   = lane >> 4;          // half-wave (0/1)
    int l16  = lane & 15;

    int m0blk = blockIdx.y * 64;
    int n0    = blockIdx.x * 64;

    // cooperative tile-load assignment: thread t -> tile row t/2, k-segment 16*(t&1)
    int trow = tid >> 1;
    int kseg = (tid & 1) << 4;
    int ar = m0blk + trow; if (ar >= Mrows) ar = Mrows - 1;   // clamp (never stored)
    int bn = n0 + trow;    if (bn >= Ncols) bn = Ncols - 1;
    long abase = aoff + (long)ar * a_rs + (long)kseg * a_cs;
    long wbase = woff + (long)bn * w_rs + (long)kseg * w_cs;
    __bf16* as_dst = &As[trow * TSTR + kseg];
    __bf16* bs_dst = &Bs[trow * TSTR + kseg];

    v8f acc[4] = {};

    for (int kk = 0; kk < Kd; kk += 32) {
        __syncthreads();   // previous step's fragment reads complete
        // ---- stage A tile (64x32 f32 -> bf16) ----
        {
            __attribute__((aligned(16))) __bf16 tmp[16];
            if (a_cs == 1) {
                const float4* p = (const float4*)(A + abase + kk);
                float4 f0 = p[0], f1 = p[1], f2 = p[2], f3 = p[3];
                tmp[0]=(__bf16)f0.x; tmp[1]=(__bf16)f0.y; tmp[2]=(__bf16)f0.z; tmp[3]=(__bf16)f0.w;
                tmp[4]=(__bf16)f1.x; tmp[5]=(__bf16)f1.y; tmp[6]=(__bf16)f1.z; tmp[7]=(__bf16)f1.w;
                tmp[8]=(__bf16)f2.x; tmp[9]=(__bf16)f2.y; tmp[10]=(__bf16)f2.z; tmp[11]=(__bf16)f2.w;
                tmp[12]=(__bf16)f3.x; tmp[13]=(__bf16)f3.y; tmp[14]=(__bf16)f3.z; tmp[15]=(__bf16)f3.w;
            } else {
                const float* p = A + abase + (long)kk * a_cs;
#pragma unroll
                for (int i = 0; i < 16; ++i) tmp[i] = (__bf16)p[(long)i * a_cs];
            }
            ((uint4*)as_dst)[0] = ((const uint4*)tmp)[0];
            ((uint4*)as_dst)[1] = ((const uint4*)tmp)[1];
        }
        // ---- stage B tile (64 cols x 32 f32 -> bf16) ----
        {
            __attribute__((aligned(16))) __bf16 tmp[16];
            if (w_cs == 1) {
                const float4* p = (const float4*)(Wt + wbase + kk);
                float4 f0 = p[0], f1 = p[1], f2 = p[2], f3 = p[3];
                tmp[0]=(__bf16)f0.x; tmp[1]=(__bf16)f0.y; tmp[2]=(__bf16)f0.z; tmp[3]=(__bf16)f0.w;
                tmp[4]=(__bf16)f1.x; tmp[5]=(__bf16)f1.y; tmp[6]=(__bf16)f1.z; tmp[7]=(__bf16)f1.w;
                tmp[8]=(__bf16)f2.x; tmp[9]=(__bf16)f2.y; tmp[10]=(__bf16)f2.z; tmp[11]=(__bf16)f2.w;
                tmp[12]=(__bf16)f3.x; tmp[13]=(__bf16)f3.y; tmp[14]=(__bf16)f3.z; tmp[15]=(__bf16)f3.w;
            } else {
                const float* p = Wt + wbase + (long)kk * w_cs;
#pragma unroll
                for (int i = 0; i < 16; ++i) tmp[i] = (__bf16)p[(long)i * w_cs];
            }
            ((uint4*)bs_dst)[0] = ((const uint4*)tmp)[0];
            ((uint4*)bs_dst)[1] = ((const uint4*)tmp)[1];
        }
        __syncthreads();

        // ---- A fragment: 16-bit A 16x32 layout (pairs; lanes16-31 K+8) ----
        const __bf16* ap = &As[(wave * 16 + l16) * TSTR];
        v16bf afr;
#pragma unroll
        for (int p = 0; p < 8; ++p) {
            int kb = ((p & 3) << 1) + ((p >> 2) << 4) + (hi << 3);
            afr[2 * p]     = ap[kb];
            afr[2 * p + 1] = ap[kb + 1];
        }
        // ---- 4 B fragments + WMMAs: 16-bit B 32x16 layout (pairs; lanes16-31 K+8, same as A)
#pragma unroll
        for (int j = 0; j < 4; ++j) {
            const __bf16* bp = &Bs[(j * 16 + l16) * TSTR + (hi << 3)];
            v16bf bfr;
#pragma unroll
            for (int p = 0; p < 8; ++p) {
                int kb = ((p & 3) << 1) + ((p >> 2) << 4);
                bfr[2 * p]     = bp[kb];
                bfr[2 * p + 1] = bp[kb + 1];
            }
            acc[j] = __builtin_amdgcn_wmma_f32_16x16x32_bf16(
                false, afr, false, bfr, (short)0, acc[j], false, false);
        }
    }

    int m0 = m0blk + wave * 16;
#pragma unroll
    for (int j = 0; j < 4; ++j) {
        int n = n0 + j * 16 + l16;
        if (n >= Ncols) continue;
        float bval = bias ? bias[n] : 0.0f;
#pragma unroll
        for (int v = 0; v < 8; ++v) {
            int row = m0 + v + (hi << 3);
            if (row < Mrows) {
                float val = acc[j][v] * alpha + bval;
                if (act == 1) val = gelu_exact(val);
                long ci = coff + (long)row * c_rs + (long)n * c_cs;
                if (resid) val += resid[ci];
                C[ci] = val;
            }
        }
    }
}

static void gemm(hipStream_t stream, const float* A, const float* W, const float* bias,
                 const float* resid, float* C, int Mr, int Nc, int Kd, int Hh, int nz,
                 long ars, long acs, long abs_, long ahs,
                 long wrs, long wcs, long wbs, long whs,
                 long crs, long ccs, long cbs, long chs, float alpha, int act)
{
    dim3 g((Nc + 63) / 64, (Mr + 63) / 64, nz);
    gemm_bf16_wmma<<<g, 128, 0, stream>>>(A, W, bias, resid, C, Mr, Nc, Kd, Hh,
                                          ars, acs, abs_, ahs, wrs, wcs, wbs, whs,
                                          crs, ccs, cbs, chs, alpha, act);
}

// ============ f0 = tl_w @ x + tl_b  -> (B,32,N) ============
__global__ void f0_kernel(const float* __restrict__ x, const float* __restrict__ w,
                          const float* __restrict__ bias, float* __restrict__ f0)
{
    int id = blockIdx.x * 256 + threadIdx.x;
    if (id >= BB * CT * NPT) return;
    int n = id & (NPT - 1);
    int o = (id >> 11) & (CT - 1);
    int b = id >> 16;
    const float* xb = x + (long)b * 3 * NPT;
    float v = bias[o];
#pragma unroll
    for (int c = 0; c < 3; ++c) v += w[o * 3 + c] * xb[c * NPT + n];
    f0[((long)b * CT + o) * NPT + n] = v;
}

// ============ brute-force kNN (top-16 smallest sq-dist, stable) ============
__global__ void knn_kernel(const float* __restrict__ q, const float* __restrict__ ref,
                           int* __restrict__ out, int Nq, int Nr, int q_bs, int ref_bs)
{
    int t = blockIdx.x * 256 + threadIdx.x;
    if (t >= BB * Nq) return;
    int b = t / Nq, qi = t % Nq;
    const float* qb = q + (long)b * q_bs;
    const float* rb = ref + (long)b * ref_bs;
    float qx = qb[qi], qy = qb[Nq + qi], qz = qb[2 * Nq + qi];
    float bd[KNN]; int bi[KNN];
#pragma unroll
    for (int i = 0; i < KNN; ++i) { bd[i] = 3.4e38f; bi[i] = 0; }
    for (int r = 0; r < Nr; ++r) {
        float dx = rb[r] - qx, dy = rb[Nr + r] - qy, dz = rb[2 * Nr + r] - qz;
        float d = dx * dx + dy * dy + dz * dz;
        if (d < bd[KNN - 1]) {
            bd[KNN - 1] = d; bi[KNN - 1] = r;
#pragma unroll
            for (int s = KNN - 1; s > 0; --s) {
                if (bd[s] < bd[s - 1]) {
                    float td = bd[s]; bd[s] = bd[s - 1]; bd[s - 1] = td;
                    int ti = bi[s];  bi[s] = bi[s - 1];  bi[s - 1] = ti;
                }
            }
        }
    }
#pragma unroll
    for (int i = 0; i < KNN; ++i) out[((long)b * Nq + qi) * KNN + i] = bi[i];
}

// ============ graph feature for one batch: gf (64, N*K) ============
__global__ void gf_kernel(const float* __restrict__ f0, const int* __restrict__ idx,
                          float* __restrict__ gf, int b)
{
    int id = blockIdx.x * 256 + threadIdx.x;   // 32*2048*16
    if (id >= CT * NPT * KNN) return;
    int k = id & 15;
    int n = (id >> 4) & (NPT - 1);
    int c = id >> 15;
    int pt = idx[((long)b * NPT + n) * KNN + k];
    float center = f0[((long)b * CT + c) * NPT + n];
    float g = f0[((long)b * CT + c) * NPT + pt];
    long sp = (long)n * KNN + k;
    gf[(long)c * (NPT * KNN) + sp]        = g - center;
    gf[(long)(c + CT) * (NPT * KNN) + sp] = center;
}

// ============ GroupNorm(groups=4 over channel*spatial) + LeakyReLU, in place ====
__global__ void gn_leaky(float* __restrict__ X, int C, int spatial,
                         const float* __restrict__ gamma, const float* __restrict__ beta)
{
    int g = blockIdx.x;           // 4 groups
    int cpg = C >> 2;
    long n = (long)cpg * spatial;
    float* base = X + (long)g * n;
    __shared__ float rs[256], rs2[256];
    float s = 0.f, s2 = 0.f;
    for (long i = threadIdx.x; i < n; i += 256) { float v = base[i]; s += v; s2 += v * v; }
    rs[threadIdx.x] = s; rs2[threadIdx.x] = s2; __syncthreads();
    for (int st = 128; st > 0; st >>= 1) {
        if ((int)threadIdx.x < st) { rs[threadIdx.x] += rs[threadIdx.x + st]; rs2[threadIdx.x] += rs2[threadIdx.x + st]; }
        __syncthreads();
    }
    float mean = rs[0] / (float)n;
    float var  = rs2[0] / (float)n - mean * mean;
    float inv  = rsqrtf(var + 1e-5f);
    for (long i = threadIdx.x; i < n; i += 256) {
        int ch = g * cpg + (int)(i / spatial);
        float v = (base[i] - mean) * inv * gamma[ch] + beta[ch];
        base[i] = (v >= 0.f) ? v : 0.2f * v;
    }
}

// ============ max over k: knn_f[b,c,n] = max_k h2[c, n*16+k] (one batch) ============
__global__ void kmax_kernel(const float* __restrict__ h2, float* __restrict__ knnf, int b)
{
    int id = blockIdx.x * 256 + threadIdx.x;   // 128*2048
    if (id >= NF1 * NPT) return;
    int n = id & (NPT - 1);
    int c = id >> 11;
    const float* p = h2 + (long)c * (NPT * KNN) + (long)n * KNN;
    float m = p[0];
#pragma unroll
    for (int k = 1; k < KNN; ++k) m = fmaxf(m, p[k]);
    knnf[((long)b * NF1 + c) * NPT + n] = m;
}

// ============ LayerNorm over last dim D (one block per row) ============
__global__ void ln_kernel(const float* __restrict__ X, const float* __restrict__ g,
                          const float* __restrict__ b, float* __restrict__ O, int D)
{
    int row = blockIdx.x;
    const float* xr = X + (long)row * D;
    float* orow = O + (long)row * D;
    __shared__ float rs[256], rs2[256];
    float s = 0.f, s2 = 0.f;
    for (int i = threadIdx.x; i < D; i += 256) { float v = xr[i]; s += v; s2 += v * v; }
    rs[threadIdx.x] = s; rs2[threadIdx.x] = s2; __syncthreads();
    for (int st = 128; st > 0; st >>= 1) {
        if ((int)threadIdx.x < st) { rs[threadIdx.x] += rs[threadIdx.x + st]; rs2[threadIdx.x] += rs2[threadIdx.x + st]; }
        __syncthreads();
    }
    float mean = rs[0] / (float)D;
    float var  = rs2[0] / (float)D - mean * mean;
    float inv  = rsqrtf(var + 1e-5f);
    for (int i = threadIdx.x; i < D; i += 256) orow[i] = (xr[i] - mean) * inv * g[i] + b[i];
}

// ============ tiny attention over t=3 tokens, one thread per (b,h) ============
__global__ void attn3_kernel(const float* __restrict__ qkv, float* __restrict__ o)
{
    int t = blockIdx.x * 64 + threadIdx.x;
    if (t >= BB * HH) return;
    int b = t >> 3, h = t & 7;
    const float* base = qkv + (long)b * 3 * (3 * DIMC) + h * DH;
    float s[3][3];
    for (int i = 0; i < 3; ++i)
        for (int j = 0; j < 3; ++j) {
            float acc = 0.f;
            for (int e = 0; e < DH; ++e)
                acc += base[i * 6144 + e] * base[2048 + j * 6144 + e];
            s[i][j] = acc * 0.0625f;
        }
    for (int i = 0; i < 3; ++i) {
        float m = fmaxf(s[i][0], fmaxf(s[i][1], s[i][2]));
        float e0 = expf(s[i][0] - m), e1 = expf(s[i][1] - m), e2 = expf(s[i][2] - m);
        float inv = 1.f / (e0 + e1 + e2);
        s[i][0] = e0 * inv; s[i][1] = e1 * inv; s[i][2] = e2 * inv;
    }
    for (int i = 0; i < 3; ++i)
        for (int e = 0; e < DH; ++e) {
            float acc = s[i][0] * base[4096 + e]
                      + s[i][1] * base[4096 + 6144 + e]
                      + s[i][2] * base[4096 + 2 * 6144 + e];
            o[(long)b * 3 * DIMC + i * DIMC + h * DH + e] = acc;
        }
}

// ============ l1 (512x3) + exact gelu ============
__global__ void l1gelu_kernel(const float* __restrict__ x1, const float* __restrict__ w,
                              float* __restrict__ hl1)
{
    int id = blockIdx.x * 256 + threadIdx.x;   // 8*512*2048
    if (id >= BB * 512 * NPT) return;
    int n = id & (NPT - 1);
    int o = (id >> 11) & 511;
    int b = id >> 20;
    float v = 0.f;
#pragma unroll
    for (int c = 0; c < 3; ++c) v += w[o * 3 + c] * x1[(long)b * 3 * NPT + c * NPT + n];
    hl1[((long)b * 512 + o) * NPT + n] = gelu_exact(v);
}

// ============ concat([xl, knn_f], last axis) ============
__global__ void concat_kernel(const float* __restrict__ xl, const float* __restrict__ knnf,
                              float* __restrict__ ccat)
{
    int id = blockIdx.x * 256 + threadIdx.x;   // 1024*4096
    if (id >= 1024 * 4096) return;
    int c = id & 4095;
    int row = id >> 12;
    float v = (c < 2048) ? xl[(long)row * 2048 + c] : knnf[(long)row * 2048 + (c - 2048)];
    ccat[(long)row * 4096 + c] = v;
}

// ============ softmax over rows of length `cols` ============
__global__ void softmax_rows(float* __restrict__ X, int rows, int cols)
{
    int r = blockIdx.x * 256 + threadIdx.x;
    if (r >= rows) return;
    float* p = X + (long)r * cols;
    float m = -3.4e38f;
    for (int i = 0; i < cols; ++i) m = fmaxf(m, p[i]);
    float s = 0.f;
    for (int i = 0; i < cols; ++i) { float e = expf(p[i] - m); p[i] = e; s += e; }
    float inv = 1.f / s;
    for (int i = 0; i < cols; ++i) p[i] *= inv;
}

// ============ farthest point sampling: one block per batch ============
__global__ void fps_kernel(const float* __restrict__ coor, int* __restrict__ idxf)
{
    int b = blockIdx.x;
    __shared__ float dist[NPT];
    __shared__ float rv[256];
    __shared__ int   ri[256];
    __shared__ int   s_last;
    __shared__ float pp[3];
    const float* cb = coor + (long)b * 3 * NPT;
    for (int i = threadIdx.x; i < NPT; i += 256) dist[i] = 1e10f;
    if (threadIdx.x == 0) { s_last = 0; idxf[b * MQ] = 0; }
    __syncthreads();
    for (int m = 1; m < MQ; ++m) {
        if (threadIdx.x == 0) {
            int l = s_last;
            pp[0] = cb[l]; pp[1] = cb[NPT + l]; pp[2] = cb[2 * NPT + l];
        }
        __syncthreads();
        float bv = -1.f; int bidx = 0;
        for (int i = threadIdx.x; i < NPT; i += 256) {
            float dx = cb[i] - pp[0], dy = cb[NPT + i] - pp[1], dz = cb[2 * NPT + i] - pp[2];
            float d = dx * dx + dy * dy + dz * dz;
            float nd = fminf(dist[i], d);
            dist[i] = nd;
            if (nd > bv) { bv = nd; bidx = i; }
        }
        rv[threadIdx.x] = bv; ri[threadIdx.x] = bidx; __syncthreads();
        for (int st = 128; st > 0; st >>= 1) {
            if ((int)threadIdx.x < st) {
                float v2 = rv[threadIdx.x + st]; int i2 = ri[threadIdx.x + st];
                if (v2 > rv[threadIdx.x] || (v2 == rv[threadIdx.x] && i2 < ri[threadIdx.x])) {
                    rv[threadIdx.x] = v2; ri[threadIdx.x] = i2;
                }
            }
            __syncthreads();
        }
        if (threadIdx.x == 0) { s_last = ri[0]; idxf[b * MQ + m] = ri[0]; }
        __syncthreads();
    }
}

// ============ gathers ============
__global__ void coorq_kernel(const float* __restrict__ x, const int* __restrict__ idxf,
                             float* __restrict__ coorq)
{
    int id = blockIdx.x * 256 + threadIdx.x;   // 8*3*1024
    if (id >= BB * 3 * MQ) return;
    int b = id / (3 * MQ);
    int rem = id % (3 * MQ);
    int c = rem / MQ, m = rem % MQ;
    coorq[((long)b * 3 + c) * MQ + m] = x[((long)b * 3 + c) * NPT + idxf[b * MQ + m]];
}

__global__ void fq_kernel(const float* __restrict__ y, const int* __restrict__ idxf,
                          float* __restrict__ fq)
{
    int id = blockIdx.x * 256 + threadIdx.x;   // 8*128*1024
    if (id >= BB * NF1 * MQ) return;
    int m = id & (MQ - 1);
    int c = (id >> 10) & (NF1 - 1);
    int b = id >> 17;
    fq[((long)b * NF1 + c) * MQ + m] = y[((long)b * NF1 + c) * NPT + idxf[b * MQ + m]];
}

__global__ void fout_kernel(const float* __restrict__ y, const float* __restrict__ fq,
                            const int* __restrict__ idx2, float* __restrict__ fo)
{
    int id = blockIdx.x * 256 + threadIdx.x;   // 8*128*1024*16 = 2^24
    if (id >= BB * NF1 * MQ * KNN) return;
    int k = id & 15;
    int m = (id >> 4) & (MQ - 1);
    int c = (id >> 14) & (NF1 - 1);
    int b = id >> 21;
    int pt = idx2[((long)b * MQ + m) * KNN + k];
    float fv = fq[((long)b * NF1 + c) * MQ + m];
    float g  = y[((long)b * NF1 + c) * NPT + pt];
    fo[(((long)b * 2 * NF1 + c) * MQ + m) * KNN + k]         = g - fv;
    fo[(((long)b * 2 * NF1 + c + NF1) * MQ + m) * KNN + k]   = fv;
}

// =====================================================================
// Host orchestration
// =====================================================================
extern "C" void kernel_launch(void* const* d_in, const int* in_sizes, int n_in,
                              void* d_out, int out_size, void* d_ws, size_t ws_size,
                              hipStream_t stream)
{
    (void)in_sizes; (void)n_in; (void)out_size; (void)ws_size;
    const float* x       = (const float*)d_in[0];
    const float* tl_w    = (const float*)d_in[1];
    const float* tl_b    = (const float*)d_in[2];
    const float* mlp_w1  = (const float*)d_in[3];
    const float* mlp_g1  = (const float*)d_in[4];
    const float* mlp_b1  = (const float*)d_in[5];
    const float* mlp_w2  = (const float*)d_in[6];
    const float* mlp_g2  = (const float*)d_in[7];
    const float* mlp_b2  = (const float*)d_in[8];
    const float* o_n1_g  = (const float*)d_in[9];
    const float* o_n1_b  = (const float*)d_in[10];
    const float* o_qkv_w = (const float*)d_in[11];
    const float* o_proj_w= (const float*)d_in[12];
    const float* o_proj_b= (const float*)d_in[13];
    const float* o_n2_g  = (const float*)d_in[14];
    const float* o_n2_b  = (const float*)d_in[15];
    const float* o_fc1_w = (const float*)d_in[16];
    const float* o_fc1_b = (const float*)d_in[17];
    const float* o_fc2_w = (const float*)d_in[18];
    const float* o_fc2_b = (const float*)d_in[19];
    const float* s_n1_g  = (const float*)d_in[20];
    const float* s_n1_b  = (const float*)d_in[21];
    const float* s_qkv_w = (const float*)d_in[22];
    const float* s_proj_w= (const float*)d_in[23];
    const float* s_proj_b= (const float*)d_in[24];
    const float* s_n2_g  = (const float*)d_in[25];
    const float* s_n2_b  = (const float*)d_in[26];
    const float* s_fc1_w = (const float*)d_in[27];
    const float* s_fc1_b = (const float*)d_in[28];
    const float* s_fc2_w = (const float*)d_in[29];
    const float* s_fc2_b = (const float*)d_in[30];
    const float* o_l1_w  = (const float*)d_in[31];
    const float* o_l2_w  = (const float*)d_in[32];
    const float* o_mg_w  = (const float*)d_in[33];
    const float* o_mg_b  = (const float*)d_in[34];

    float* ws = (float*)d_ws;
    // workspace layout (float offsets)
    constexpr long O_F0    = 0;                       // 8*32*2048      = 524288
    constexpr long O_GF    = O_F0    + 524288;        // 64*32768       = 2097152 (per batch)
    constexpr long O_H1    = O_GF    + 2097152;       // 256*32768      = 8388608 (per batch)
    constexpr long O_H2    = O_H1    + 8388608;       // 128*32768      = 4194304 (per batch)
    constexpr long O_KNNF  = O_H2    + 4194304;       // 8*128*2048     = 2097152
    constexpr long O_XLN   = O_KNNF  + 2097152;       // 24*2048        = 49152
    constexpr long O_QKV1  = O_XLN   + 49152;         // 24*6144        = 147456
    constexpr long O_ATT1  = O_QKV1  + 147456;        // 24*2048
    constexpr long O_X1    = O_ATT1  + 49152;         // 24*2048
    constexpr long O_HL1   = O_X1    + 49152;         // 8*512*2048     = 8388608
    constexpr long O_XL    = O_HL1   + 8388608;       // 8*128*2048     = 2097152
    constexpr long O_CCAT  = O_XL    + 2097152;       // 1024*4096      = 4194304
    constexpr long O_Y     = O_CCAT  + 4194304;       // 1024*2048      = 2097152
    constexpr long O_LNB   = O_Y     + 2097152;       // 1024*2048
    constexpr long O_FC1O  = O_LNB   + 2097152;       // 1024*4096      = 4194304
    constexpr long O_QKVY  = O_FC1O  + 4194304;       // 1024*6144      = 6291456
    constexpr long O_SCOR  = O_QKVY  + 6291456;       // 64*128*128     = 1048576
    constexpr long O_ATTO  = O_SCOR  + 1048576;       // 1024*2048      = 2097152
    constexpr long O_IDX1  = O_ATTO  + 2097152;       // ints: 8*2048*16
    constexpr long O_IDXF  = O_IDX1  + 262144;        // ints: 8*1024
    constexpr long O_IDX2  = O_IDXF  + 8192;          // ints: 8*1024*16

    float* f0   = ws + O_F0;
    float* gf   = ws + O_GF;
    float* h1   = ws + O_H1;
    float* h2   = ws + O_H2;
    float* knnf = ws + O_KNNF;
    float* xln  = ws + O_XLN;
    float* qkv1 = ws + O_QKV1;
    float* att1 = ws + O_ATT1;
    float* x1b  = ws + O_X1;
    float* hl1  = ws + O_HL1;
    float* xl   = ws + O_XL;
    float* ccat = ws + O_CCAT;
    float* Y    = ws + O_Y;
    float* lnb  = ws + O_LNB;
    float* fc1o = ws + O_FC1O;
    float* qkvy = ws + O_QKVY;
    float* scor = ws + O_SCOR;
    float* atto = ws + O_ATTO;
    int*   idx1 = (int*)(ws + O_IDX1);
    int*   idxf = (int*)(ws + O_IDXF);
    int*   idx2 = (int*)(ws + O_IDX2);

    float* out   = (float*)d_out;
    float* coorq = out;                       // 8*3*1024
    float* fqo   = out + 24576;               // 8*128*1024
    float* fouto = out + 24576 + 1048576;     // 8*256*1024*16

    // ---- token embedding + self-kNN ----
    f0_kernel<<<(BB * CT * NPT + 255) / 256, 256, 0, stream>>>(x, tl_w, tl_b, f0);
    knn_kernel<<<(BB * NPT + 255) / 256, 256, 0, stream>>>(x, x, idx1, NPT, NPT, 3 * NPT, 3 * NPT);

    // ---- DGCNN edge-conv, one batch at a time (saves workspace) ----
    for (int b = 0; b < BB; ++b) {
        gf_kernel<<<(CT * NPT * KNN + 255) / 256, 256, 0, stream>>>(f0, idx1, gf, b);
        // h1 = mlp_w1 (256x64) applied to gf columns; rows = n*k = 32768
        gemm(stream, gf, mlp_w1, nullptr, nullptr, h1, 32768, 256, 64, 1, 1,
             1, 32768, 0, 0,   64, 1, 0, 0,   1, 32768, 0, 0,   1.0f, 0);
        gn_leaky<<<4, 256, 0, stream>>>(h1, 256, 32768, mlp_g1, mlp_b1);
        gemm(stream, h1, mlp_w2, nullptr, nullptr, h2, 32768, 128, 256, 1, 1,
             1, 32768, 0, 0,   256, 1, 0, 0,   1, 32768, 0, 0,   1.0f, 0);
        gn_leaky<<<4, 256, 0, stream>>>(h2, 128, 32768, mlp_g2, mlp_b2);
        kmax_kernel<<<(NF1 * NPT + 255) / 256, 256, 0, stream>>>(h2, knnf, b);
    }

    // ---- attention over t=3 on x ----
    ln_kernel<<<BB * 3, 256, 0, stream>>>(x, o_n1_g, o_n1_b, xln, DIMC);
    gemm(stream, xln, o_qkv_w, nullptr, nullptr, qkv1, 24, 3 * DIMC, DIMC, 1, 1,
         DIMC, 1, 0, 0,   DIMC, 1, 0, 0,   3 * DIMC, 1, 0, 0,   1.0f, 0);
    attn3_kernel<<<1, 64, 0, stream>>>(qkv1, att1);
    gemm(stream, att1, o_proj_w, o_proj_b, nullptr, x1b, 24, DIMC, DIMC, 1, 1,
         DIMC, 1, 0, 0,   DIMC, 1, 0, 0,   DIMC, 1, 0, 0,   1.0f, 0);

    // ---- xl = l2( gelu(l1(x1)) ) ----
    l1gelu_kernel<<<(BB * 512 * NPT + 255) / 256, 256, 0, stream>>>(x1b, o_l1_w, hl1);
    gemm(stream, hl1, o_l2_w, nullptr, nullptr, xl, NPT, NF1, 512, 1, BB,
         1, NPT, 512L * NPT, 0,   512, 1, 0, 0,   1, NPT, (long)NF1 * NPT, 0,   1.0f, 0);

    // ---- y = xl + (concat([xl,knn_f]) @ mg_w^T + mg_b) ----
    concat_kernel<<<(1024 * 4096 + 255) / 256, 256, 0, stream>>>(xl, knnf, ccat);
    gemm(stream, ccat, o_mg_w, o_mg_b, xl, Y, 1024, DIMC, 2 * DIMC, 1, 1,
         2 * DIMC, 1, 0, 0,   2 * DIMC, 1, 0, 0,   DIMC, 1, 0, 0,   1.0f, 0);

    // ---- y += MLP(LN(y))  (o block) ----
    ln_kernel<<<1024, 256, 0, stream>>>(Y, o_n2_g, o_n2_b, lnb, DIMC);
    gemm(stream, lnb, o_fc1_w, o_fc1_b, nullptr, fc1o, 1024, HIDN, DIMC, 1, 1,
         DIMC, 1, 0, 0,   DIMC, 1, 0, 0,   HIDN, 1, 0, 0,   1.0f, 1);
    gemm(stream, fc1o, o_fc2_w, o_fc2_b, Y, Y, 1024, DIMC, HIDN, 1, 1,
         HIDN, 1, 0, 0,   HIDN, 1, 0, 0,   DIMC, 1, 0, 0,   1.0f, 0);

    // ---- y += Attn(LN(y))  (s block); t=128, heads=8, dh=256 ----
    ln_kernel<<<1024, 256, 0, stream>>>(Y, s_n1_g, s_n1_b, lnb, DIMC);
    gemm(stream, lnb, s_qkv_w, nullptr, nullptr, qkvy, 1024, 3 * DIMC, DIMC, 1, 1,
         DIMC, 1, 0, 0,   DIMC, 1, 0, 0,   3 * DIMC, 1, 0, 0,   1.0f, 0);
    // scores[b,h] = (q @ k^T) * dh^-0.5   — batched over z = b*8+h
    gemm(stream, qkvy, qkvy + 2048, nullptr, nullptr, scor, TT, TT, DH, HH, 64,
         6144, 1, 786432, 256,   6144, 1, 786432, 256,   128, 1, 131072, 16384,
         0.0625f, 0);
    softmax_rows<<<(64 * TT + 255) / 256, 256, 0, stream>>>(scor, 64 * TT, TT);
    // attno[b,t,h*dh+e] = probs @ v
    gemm(stream, scor, qkvy + 4096, nullptr, nullptr, atto, TT, DH, TT, HH, 64,
         128, 1, 131072, 16384,   1, 6144, 786432, 256,   2048, 1, 262144, 256,
         1.0f, 0);
    gemm(stream, atto, s_proj_w, s_proj_b, Y, Y, 1024, DIMC, DIMC, 1, 1,
         DIMC, 1, 0, 0,   DIMC, 1, 0, 0,   DIMC, 1, 0, 0,   1.0f, 0);

    // ---- y += MLP(LN(y))  (s block) ----
    ln_kernel<<<1024, 256, 0, stream>>>(Y, s_n2_g, s_n2_b, lnb, DIMC);
    gemm(stream, lnb, s_fc1_w, s_fc1_b, nullptr, fc1o, 1024, HIDN, DIMC, 1, 1,
         DIMC, 1, 0, 0,   DIMC, 1, 0, 0,   HIDN, 1, 0, 0,   1.0f, 1);
    gemm(stream, fc1o, s_fc2_w, s_fc2_b, Y, Y, 1024, DIMC, HIDN, 1, 1,
         HIDN, 1, 0, 0,   HIDN, 1, 0, 0,   DIMC, 1, 0, 0,   1.0f, 0);

    // ---- FPS + output gathers ----
    fps_kernel<<<BB, 256, 0, stream>>>(x, idxf);
    coorq_kernel<<<(BB * 3 * MQ + 255) / 256, 256, 0, stream>>>(x, idxf, coorq);
    knn_kernel<<<(BB * MQ + 255) / 256, 256, 0, stream>>>(coorq, x, idx2, MQ, NPT, 3 * MQ, 3 * NPT);
    fq_kernel<<<(BB * NF1 * MQ + 255) / 256, 256, 0, stream>>>(Y, idxf, fqo);
    fout_kernel<<<(BB * NF1 * MQ * KNN + 255) / 256, 256, 0, stream>>>(Y, fqo, idx2, fouto);
}
